// myModel_ICDM_77953656422623
// MI455X (gfx1250) — matrix operand order repaired and checked by reading, not compiled, not run on hardware
//
#include <hip/hip_runtime.h>

// ---------------------------------------------------------------------------
// Types for CDNA5 WMMA
// ---------------------------------------------------------------------------
typedef __attribute__((ext_vector_type(16))) __bf16 bf16x16;
typedef __attribute__((ext_vector_type(8)))  float  floatx8;

// LDS tile geometry (bytes). A tile: 128 rows x 32 bf16 (64B) + 16B pad.
// B tile: 32 rows x 128 bf16 (256B) + 16B pad.
#define LDSA_PITCH 80
#define LDSB_PITCH 272
#define LDSA_SZ (128 * LDSA_PITCH)          // 10240
#define LDSB_SZ (32 * LDSB_PITCH)           // 8704
#define LDS_BUF (LDSA_SZ + LDSB_SZ)         // 18944

// Pooling kernel B tile: 32 rows x 64 bf16 (128B) + 16B pad.
#define PB_PITCH 144
#define PB_SZ (32 * PB_PITCH)               // 4608
#define PLDS_BUF (LDSA_SZ + PB_SZ)          // 14848

// ---------------------------------------------------------------------------
// CDNA5 async global->LDS copies (guarded: fall back to sync copy if the
// toolchain lacks the builtins). The builtin's params are pointers to a
// 128-bit vector type, global AS for src and LDS AS for dst.
// ---------------------------------------------------------------------------
#if __has_builtin(__builtin_amdgcn_global_load_async_to_lds_b128)
#define HAS_ASYNC 1
#define AS1 __attribute__((address_space(1)))
#define AS3 __attribute__((address_space(3)))
typedef int b128_t __attribute__((vector_size(16)));
__device__ __forceinline__ void async_g2l_b128(const void* g, void* l) {
  __builtin_amdgcn_global_load_async_to_lds_b128(
      (AS1 b128_t*)(void*)g, (AS3 b128_t*)l, 0, 0);
}
#else
#define HAS_ASYNC 0
#endif

__device__ __forceinline__ void async_fence() {
#if HAS_ASYNC
#if __has_builtin(__builtin_amdgcn_s_wait_asynccnt)
  __builtin_amdgcn_s_wait_asynccnt(0);
#else
  asm volatile("s_wait_asynccnt 0x0" ::: "memory");
#endif
#endif
}

// ---------------------------------------------------------------------------
// fp32 -> bf16 helpers: use the native cast so the backend can select the
// gfx1250 HW convert (RTNE either way).
// ---------------------------------------------------------------------------
__device__ __forceinline__ unsigned short f2bf(float x) {
  __bf16 h = (__bf16)x;
  return __builtin_bit_cast(unsigned short, h);
}
typedef __attribute__((ext_vector_type(2))) __bf16 bf16x2;
__device__ __forceinline__ unsigned pk2(float a, float b) {
  bf16x2 v;
  v.x = (__bf16)a;
  v.y = (__bf16)b;
  return __builtin_bit_cast(unsigned, v);
}

__device__ __forceinline__ bf16x16 frag2(const unsigned char* p0,
                                         const unsigned char* p1) {
  union { uint4 u[2]; bf16x16 v; } f;
  f.u[0] = *(const uint4*)p0;
  f.u[1] = *(const uint4*)p1;
  return f.v;
}

// ---------------------------------------------------------------------------
// Generic GEMM:  C = act(A[M,K] @ B[K,N] + bias)
//   A: fp32 (converted to bf16 on the fly) or bf16, row-major, leading dim lda
//   B: bf16 row-major [K,N], leading dim ldb
//   Outputs: optional fp32 C (ldcf) and optional bf16 C (ldcb)
// Block: 256 threads = 8 waves; block tile 128x128; BK=32; wave tile 32x64.
// ---------------------------------------------------------------------------
template <bool ABF16, bool RELU>
__global__ __launch_bounds__(256) void gemm_wmma(
    const void* __restrict__ Ap, const unsigned short* __restrict__ B,
    const float* __restrict__ bias, int K, int lda, int ldb,
    float* __restrict__ Cf, int ldcf, unsigned short* __restrict__ Cb,
    int ldcb) {
  __shared__ alignas(16) unsigned char smem[2 * LDS_BUF];
  const int t = threadIdx.x;
  const int lane = t & 31;
  const int w = t >> 5;
  const int wm = w & 3;          // 4 waves along M
  const int wn = w >> 2;         // 2 waves along N
  const int m0 = blockIdx.y * 128;
  const int n0 = blockIdx.x * 128;

  floatx8 acc[2][4];
  const floatx8 zero = {0.f, 0.f, 0.f, 0.f, 0.f, 0.f, 0.f, 0.f};
#pragma unroll
  for (int i = 0; i < 2; ++i)
#pragma unroll
    for (int j = 0; j < 4; ++j) acc[i][j] = zero;

  const int arow = t >> 1;       // 0..127
  const int ahalf = t & 1;       // which 16-col half of the 32-wide K slab

  auto loadTile = [&](int buf, int k0) {
    unsigned char* la = smem + buf * LDS_BUF;
    unsigned char* lb = la + LDSA_SZ;
    // ---- A tile: 128 x 32 -> LDS [row][k] bf16
    if constexpr (!ABF16) {
      const float* ga =
          (const float*)Ap + (size_t)(m0 + arow) * lda + k0 + ahalf * 16;
      float4 f0 = ((const float4*)ga)[0];
      float4 f1 = ((const float4*)ga)[1];
      float4 f2 = ((const float4*)ga)[2];
      float4 f3 = ((const float4*)ga)[3];
      uint4 u0, u1;
      u0.x = pk2(f0.x, f0.y); u0.y = pk2(f0.z, f0.w);
      u0.z = pk2(f1.x, f1.y); u0.w = pk2(f1.z, f1.w);
      u1.x = pk2(f2.x, f2.y); u1.y = pk2(f2.z, f2.w);
      u1.z = pk2(f3.x, f3.y); u1.w = pk2(f3.z, f3.w);
      unsigned char* d = la + arow * LDSA_PITCH + ahalf * 32;
      *(uint4*)d = u0;
      *(uint4*)(d + 16) = u1;
    } else {
      const unsigned short* ga =
          (const unsigned short*)Ap + (size_t)(m0 + arow) * lda + k0 +
          ahalf * 16;
      unsigned char* d = la + arow * LDSA_PITCH + ahalf * 32;
#if HAS_ASYNC
      async_g2l_b128(ga, d);
      async_g2l_b128(ga + 8, d + 16);
#else
      uint4 u0 = ((const uint4*)ga)[0];
      uint4 u1 = ((const uint4*)ga)[1];
      *(uint4*)d = u0;
      *(uint4*)(d + 16) = u1;
#endif
    }
    // ---- B tile: 32 x 128 bf16 -> LDS [k][n]
#pragma unroll
    for (int j = 0; j < 2; ++j) {
      int idx = t + j * 256;                 // 0..511 uint4 groups
      int br = idx >> 4;                     // 0..31
      int bc = (idx & 15) * 8;               // 0..120
      const unsigned short* gb = B + (size_t)(k0 + br) * ldb + n0 + bc;
      unsigned char* d = lb + br * LDSB_PITCH + bc * 2;
#if HAS_ASYNC
      async_g2l_b128(gb, d);
#else
      *(uint4*)d = *(const uint4*)gb;
#endif
    }
  };

  const int nK = K >> 5;
  loadTile(0, 0);
  async_fence();
  __syncthreads();
  for (int kb = 0; kb < nK; ++kb) {
    const int cur = kb & 1;
    if (kb + 1 < nK) loadTile(cur ^ 1, (kb + 1) << 5);
    const unsigned char* la = smem + cur * LDS_BUF;
    const unsigned char* lb = la + LDSA_SZ;
    bf16x16 af[2], bfr[4];
#pragma unroll
    for (int mi = 0; mi < 2; ++mi) {
      const unsigned char* p =
          la + (wm * 32 + mi * 16 + (lane & 15)) * LDSA_PITCH +
          ((lane >> 4) * 16);
      af[mi] = frag2(p, p + 32);
    }
#pragma unroll
    for (int ni = 0; ni < 4; ++ni) {
      const unsigned char* p = lb + (lane >> 4) * 8 * LDSB_PITCH + (wn * 64 + ni * 16 + (lane & 15)) * 2;
#pragma unroll
      for (int s = 0; s < 16; ++s)
        bfr[ni][s] = *(const __bf16*)(p + ((s & 7) + (s >> 3) * 16) * LDSB_PITCH);
    }
#pragma unroll
    for (int mi = 0; mi < 2; ++mi)
#pragma unroll
      for (int ni = 0; ni < 4; ++ni)
        acc[mi][ni] = __builtin_amdgcn_wmma_f32_16x16x32_bf16(
            false, af[mi], false, bfr[ni], (short)0, acc[mi][ni], false, false);
    async_fence();
    __syncthreads();
  }

  // ---- epilogue: bias + relu + stores
  const int nIdx = lane & 15;
  const int mHi = (lane >> 4) * 8;
#pragma unroll
  for (int ni = 0; ni < 4; ++ni) {
    const int gn = n0 + wn * 64 + ni * 16 + nIdx;
    const float bv = bias ? bias[gn] : 0.f;
#pragma unroll
    for (int mi = 0; mi < 2; ++mi) {
#pragma unroll
      for (int v = 0; v < 8; ++v) {
        const int gm = m0 + wm * 32 + mi * 16 + mHi + v;
        float x = acc[mi][ni][v] + bv;
        if (RELU) x = fmaxf(x, 0.f);
        if (Cf) Cf[(size_t)gm * ldcf + gn] = x;
        if (Cb) Cb[(size_t)gm * ldcb + gn] = f2bf(x);
      }
    }
  }
}

// ---------------------------------------------------------------------------
// Pooling GEMM:  C[i,n] = sum_k H[k,i] * Hm[k,n]   (A operand is H^T)
//   H : fp32 [K, 4096] row-major (transpose + cvt on LDS store)
//   Hm: bf16 [K, 512] row-major
//   Out: bf16, leading dim ldo (writes the sum half of the Event concat)
// Block tile 128(i) x 64(n), BK=32; 8 waves, wave tile 32x32.
// ---------------------------------------------------------------------------
__global__ __launch_bounds__(256) void pool_wmma(
    const float* __restrict__ H, const unsigned short* __restrict__ Hm, int K,
    int ldh, int ldm, unsigned short* __restrict__ Out, int ldo) {
  __shared__ alignas(16) unsigned char smem[2 * PLDS_BUF];
  const int t = threadIdx.x;
  const int lane = t & 31;
  const int w = t >> 5;
  const int wm = w & 3;          // 4 waves along i
  const int wn = w >> 2;         // 2 waves along n
  const int i0 = blockIdx.y * 128;
  const int n0 = blockIdx.x * 64;

  floatx8 acc[2][2];
  const floatx8 zero = {0.f, 0.f, 0.f, 0.f, 0.f, 0.f, 0.f, 0.f};
#pragma unroll
  for (int i = 0; i < 2; ++i)
#pragma unroll
    for (int j = 0; j < 2; ++j) acc[i][j] = zero;

  // Each thread: two adjacent k rows (kp, kp+1) x 8 consecutive i columns.
  const int kp = (t & 15) * 2;   // 0,2,...,30
  const int ig = (t >> 4) * 8;   // 0,8,...,120

  auto loadTile = [&](int buf, int k0) {
    unsigned char* la = smem + buf * PLDS_BUF;
    unsigned char* lb = la + LDSA_SZ;
    // A' tile: LDS [i][k] <- H[k][i]; 4 coalesced loads first, then pack
    // (k,k+1) pairs into dword LDS stores.
    const float* r0 = H + (size_t)(k0 + kp) * ldh + i0 + ig;
    const float* r1 = H + (size_t)(k0 + kp + 1) * ldh + i0 + ig;
    float4 a0 = ((const float4*)r0)[0];
    float4 a1 = ((const float4*)r0)[1];
    float4 b0 = ((const float4*)r1)[0];
    float4 b1 = ((const float4*)r1)[1];
    unsigned char* base = la + kp * 2;
    *(unsigned*)(base + (ig + 0) * LDSA_PITCH) = pk2(a0.x, b0.x);
    *(unsigned*)(base + (ig + 1) * LDSA_PITCH) = pk2(a0.y, b0.y);
    *(unsigned*)(base + (ig + 2) * LDSA_PITCH) = pk2(a0.z, b0.z);
    *(unsigned*)(base + (ig + 3) * LDSA_PITCH) = pk2(a0.w, b0.w);
    *(unsigned*)(base + (ig + 4) * LDSA_PITCH) = pk2(a1.x, b1.x);
    *(unsigned*)(base + (ig + 5) * LDSA_PITCH) = pk2(a1.y, b1.y);
    *(unsigned*)(base + (ig + 6) * LDSA_PITCH) = pk2(a1.z, b1.z);
    *(unsigned*)(base + (ig + 7) * LDSA_PITCH) = pk2(a1.w, b1.w);
    // B tile: LDS [k][n] <- Hm[k][n], straight bf16 copy
    {
      int br = t >> 3;                        // 0..31
      int bc = (t & 7) * 8;                   // 0..56
      const unsigned short* gb = Hm + (size_t)(k0 + br) * ldm + n0 + bc;
      unsigned char* d = lb + br * PB_PITCH + bc * 2;
#if HAS_ASYNC
      async_g2l_b128(gb, d);
#else
      *(uint4*)d = *(const uint4*)gb;
#endif
    }
  };

  const int nK = K >> 5;
  loadTile(0, 0);
  async_fence();
  __syncthreads();
  for (int kb = 0; kb < nK; ++kb) {
    const int cur = kb & 1;
    if (kb + 1 < nK) loadTile(cur ^ 1, (kb + 1) << 5);
    const unsigned char* la = smem + cur * PLDS_BUF;
    const unsigned char* lb = la + LDSA_SZ;
    bf16x16 af[2], bfr[2];
#pragma unroll
    for (int mi = 0; mi < 2; ++mi) {
      const unsigned char* p =
          la + (wm * 32 + mi * 16 + (lane & 15)) * LDSA_PITCH +
          ((lane >> 4) * 16);
      af[mi] = frag2(p, p + 32);
    }
#pragma unroll
    for (int ni = 0; ni < 2; ++ni) {
      const unsigned char* p = lb + (lane >> 4) * 8 * PB_PITCH + (wn * 32 + ni * 16 + (lane & 15)) * 2;
#pragma unroll
      for (int s = 0; s < 16; ++s)
        bfr[ni][s] = *(const __bf16*)(p + ((s & 7) + (s >> 3) * 16) * PB_PITCH);
    }
#pragma unroll
    for (int mi = 0; mi < 2; ++mi)
#pragma unroll
      for (int ni = 0; ni < 2; ++ni)
        acc[mi][ni] = __builtin_amdgcn_wmma_f32_16x16x32_bf16(
            false, af[mi], false, bfr[ni], (short)0, acc[mi][ni], false, false);
    async_fence();
    __syncthreads();
  }

  const int nIdx = lane & 15;
  const int mHi = (lane >> 4) * 8;
#pragma unroll
  for (int ni = 0; ni < 2; ++ni) {
    const int gn = n0 + wn * 32 + ni * 16 + nIdx;
#pragma unroll
    for (int mi = 0; mi < 2; ++mi) {
#pragma unroll
      for (int v = 0; v < 8; ++v) {
        const int gi = i0 + wm * 32 + mi * 16 + mHi + v;
        Out[(size_t)gi * ldo + gn] = f2bf(acc[mi][ni][v]);
      }
    }
  }
}

// ---------------------------------------------------------------------------
// fp32 -> bf16 elementwise (n must be a multiple of 2048; grid = n/2048)
// ---------------------------------------------------------------------------
__global__ __launch_bounds__(256) void cvt_bf16(const float* __restrict__ in,
                                                unsigned short* __restrict__ out,
                                                size_t n) {
  size_t i = ((size_t)blockIdx.x * 256 + threadIdx.x) * 8;
  if (i >= n) return;
  float4 a = *(const float4*)(in + i);
  float4 b = *(const float4*)(in + i + 4);
  uint4 u;
  u.x = pk2(a.x, a.y);
  u.y = pk2(a.z, a.w);
  u.z = pk2(b.x, b.y);
  u.w = pk2(b.z, b.w);
  *(uint4*)(out + i) = u;
}

// Copy Event[:, :512] (bf16, ld=1024) from src to dst. Grid = 1024 x 256.
__global__ __launch_bounds__(256) void copy_event_half(
    const unsigned short* __restrict__ src, unsigned short* __restrict__ dst) {
  int idx = blockIdx.x * 256 + threadIdx.x;  // one uint4 (8 bf16) each
  int row = idx >> 6;                        // 64 uint4 per 512-col half-row
  int col = (idx & 63) * 8;
  *(uint4*)(dst + (size_t)row * 1024 + col) =
      *(const uint4*)(src + (size_t)row * 1024 + col);
}

// ---------------------------------------------------------------------------
// Host-side orchestration
// ---------------------------------------------------------------------------
extern "C" void kernel_launch(void* const* d_in, const int* in_sizes, int n_in,
                              void* d_out, int out_size, void* d_ws,
                              size_t ws_size, hipStream_t stream) {
  const float* H_A = (const float*)d_in[0];
  const float* H_D = (const float*)d_in[1];
  const float* H_M = (const float*)d_in[2];
  const float* wA1 = (const float*)d_in[3];
  const float* bA1 = (const float*)d_in[4];
  const float* wA2 = (const float*)d_in[5];
  const float* bA2 = (const float*)d_in[6];
  const float* wD1 = (const float*)d_in[7];
  const float* bD1 = (const float*)d_in[8];
  const float* wD2 = (const float*)d_in[9];
  const float* bD2 = (const float*)d_in[10];
  const float* wM1 = (const float*)d_in[11];
  const float* bM1 = (const float*)d_in[12];
  const float* wM2 = (const float*)d_in[13];
  const float* bM2 = (const float*)d_in[14];
  const float* wEA1 = (const float*)d_in[15];
  const float* bEA1 = (const float*)d_in[16];
  const float* wEA2 = (const float*)d_in[17];
  const float* bEA2 = (const float*)d_in[18];
  const float* wED1 = (const float*)d_in[19];
  const float* bED1 = (const float*)d_in[20];
  const float* wED2 = (const float*)d_in[21];
  const float* bED2 = (const float*)d_in[22];

  const int NA = 16384, DEV = 4096, H1 = 1024, HD = 512, H3 = 1024, DM = 4096;

  float* out = (float*)d_out;
  float* recA = out;
  float* recD = recA + (size_t)DEV * DM;
  float* HmAo = recD + (size_t)DEV * DM;
  float* HmDo = HmAo + (size_t)NA * HD;
  float* HmMo = HmDo + (size_t)NA * HD;

  unsigned short* ws = (unsigned short*)d_ws;
  size_t off = 0;
  auto alloc = [&](size_t n) {
    unsigned short* p = ws + off;
    off += n;
    return p;
  };
  unsigned short* wA1b = alloc((size_t)DEV * H1);
  unsigned short* wA2b = alloc((size_t)H1 * HD);
  unsigned short* wD1b = alloc((size_t)DEV * H1);
  unsigned short* wD2b = alloc((size_t)H1 * HD);
  unsigned short* wM1b = alloc((size_t)DEV * H1);
  unsigned short* wM2b = alloc((size_t)H1 * HD);
  unsigned short* wEA1b = alloc((size_t)(2 * HD) * H3);
  unsigned short* wEA2b = alloc((size_t)H3 * DM);
  unsigned short* wED1b = alloc((size_t)(2 * HD) * H3);
  unsigned short* wED2b = alloc((size_t)H3 * DM);
  unsigned short* H1A = alloc((size_t)NA * H1);
  unsigned short* H1D = alloc((size_t)NA * H1);
  unsigned short* H1M = alloc((size_t)DEV * H1);
  unsigned short* HmAb = alloc((size_t)NA * HD);
  unsigned short* HmDb = alloc((size_t)NA * HD);
  unsigned short* EvA = alloc((size_t)DEV * (2 * HD));
  unsigned short* EvD = alloc((size_t)DEV * (2 * HD));
  unsigned short* RAh = alloc((size_t)DEV * H3);
  unsigned short* RDh = alloc((size_t)DEV * H3);

  // 1) weights -> bf16
  auto cvt = [&](const float* src, unsigned short* dst, size_t n) {
    cvt_bf16<<<dim3((unsigned)(n / 2048)), dim3(256), 0, stream>>>(src, dst, n);
  };
  cvt(wA1, wA1b, (size_t)DEV * H1);
  cvt(wA2, wA2b, (size_t)H1 * HD);
  cvt(wD1, wD1b, (size_t)DEV * H1);
  cvt(wD2, wD2b, (size_t)H1 * HD);
  cvt(wM1, wM1b, (size_t)DEV * H1);
  cvt(wM2, wM2b, (size_t)H1 * HD);
  cvt(wEA1, wEA1b, (size_t)(2 * HD) * H3);
  cvt(wEA2, wEA2b, (size_t)H3 * DM);
  cvt(wED1, wED1b, (size_t)(2 * HD) * H3);
  cvt(wED2, wED2b, (size_t)H3 * DM);

  // 2) encoder layer 1 (fp32 A, relu, bf16 out)
  gemm_wmma<false, true><<<dim3(H1 / 128, NA / 128), 256, 0, stream>>>(
      H_A, wA1b, bA1, DEV, DEV, H1, nullptr, 0, H1A, H1);
  gemm_wmma<false, true><<<dim3(H1 / 128, NA / 128), 256, 0, stream>>>(
      H_D, wD1b, bD1, DEV, DEV, H1, nullptr, 0, H1D, H1);
  gemm_wmma<false, true><<<dim3(H1 / 128, DEV / 128), 256, 0, stream>>>(
      H_M, wM1b, bM1, DEV, DEV, H1, nullptr, 0, H1M, H1);

  // 3) encoder layer 2 (bf16 A, no relu, fp32 out + bf16 copy)
  gemm_wmma<true, false><<<dim3(HD / 128, NA / 128), 256, 0, stream>>>(
      H1A, wA2b, bA2, H1, H1, HD, HmAo, HD, HmAb, HD);
  gemm_wmma<true, false><<<dim3(HD / 128, NA / 128), 256, 0, stream>>>(
      H1D, wD2b, bD2, H1, H1, HD, HmDo, HD, HmDb, HD);
  // M: bf16 copy goes straight into EventA columns [0,512)
  gemm_wmma<true, false><<<dim3(HD / 128, DEV / 128), 256, 0, stream>>>(
      H1M, wM2b, bM2, H1, H1, HD, HmMo, HD, EvA, 2 * HD);

  // 4) EventD[:, :512] = EventA[:, :512] (= H_mean_M bf16)
  copy_event_half<<<dim3(1024), dim3(256), 0, stream>>>(EvA, EvD);

  // 5) pooling: Event[:, 512:] = H^T @ H_mean
  pool_wmma<<<dim3(HD / 64, DEV / 128), 256, 0, stream>>>(
      H_A, HmAb, NA, DEV, HD, EvA + HD, 2 * HD);
  pool_wmma<<<dim3(HD / 64, DEV / 128), 256, 0, stream>>>(
      H_D, HmDb, NA, DEV, HD, EvD + HD, 2 * HD);

  // 6) reconstruction MLPs
  gemm_wmma<true, true><<<dim3(H3 / 128, DEV / 128), 256, 0, stream>>>(
      EvA, wEA1b, bEA1, 2 * HD, 2 * HD, H3, nullptr, 0, RAh, H3);
  gemm_wmma<true, true><<<dim3(H3 / 128, DEV / 128), 256, 0, stream>>>(
      EvD, wED1b, bED1, 2 * HD, 2 * HD, H3, nullptr, 0, RDh, H3);
  gemm_wmma<true, false><<<dim3(DM / 128, DEV / 128), 256, 0, stream>>>(
      RAh, wEA2b, bEA2, H3, H3, DM, recA, DM, nullptr, 0);
  gemm_wmma<true, false><<<dim3(DM / 128, DEV / 128), 256, 0, stream>>>(
      RDh, wED2b, bED2, H3, H3, DM, recD, DM, nullptr, 0);
}
